// SemanticBlock_30408368456176
// MI455X (gfx1250) — matrix operand order repaired and checked by reading, not compiled, not run on hardware
//
#include <hip/hip_runtime.h>
#include <hip/hip_bf16.h>

typedef __attribute__((ext_vector_type(16))) _Float16 v16h;
typedef __attribute__((ext_vector_type(8)))  float    v8f;

#define EPS 1e-5f

// ---- WMMA per-lane layout helpers (V_WMMA_F32_16X16X32_F16, wave32) ----
// A (16x32 f16): lane L holds row M=L&15; element e -> K = e + (L>=16 ? 16 : 0)
__device__ __forceinline__ int akidx(int e, int lane) {
    return e + (lane & 16);
}
// B (32x16 f16): lane L holds col N=L&15; element e -> K = e + (L>=16 ? 16 : 0)
__device__ __forceinline__ int bkidx(int e, int lane) { return e + (lane & 16); }
// C/D (16x16 f32): lane L holds col N=L&15, rows M = vg + (L>=16 ? 8 : 0)

__device__ __forceinline__ v16h vz16() {
    v16h z;
#pragma unroll
    for (int i = 0; i < 16; ++i) z[i] = (_Float16)0.f;
    return z;
}
__device__ __forceinline__ v8f vz8() {
    v8f z;
#pragma unroll
    for (int i = 0; i < 8; ++i) z[i] = 0.f;
    return z;
}
__device__ __forceinline__ v8f wmma32(v16h a, v16h b, v8f c) {
    return __builtin_amdgcn_wmma_f32_16x16x32_f16(false, a, false, b, (short)0, c, false, false);
}

// ---------------- problem constants (from reference setup) ----------------
// B=4, N=4096, D=1024, K=48, H=W=64, Cmid=128

// ============ weight packing into WMMA B-layout (f16) ============
// dst flat: ((ks*12 + nt)*32 + lane)*16 + e ; nt 0..11 covers [skip|q|k|v] cols
__global__ void pack_proj_kernel(const float* __restrict__ sw, const float* __restrict__ qw,
                                 const float* __restrict__ kw2, const float* __restrict__ vw,
                                 _Float16* __restrict__ dst) {
    int idx = blockIdx.x * 256 + threadIdx.x;
    if (idx >= 32 * 12 * 512) return;
    int e = idx & 15, lane = (idx >> 4) & 31;
    int rest = idx >> 9;
    int nt = rest % 12, ks = rest / 12;
    int K = bkidx(e, lane);
    int c = ks * 32 + K;              // D index, always < 1024
    int j = nt * 16 + (lane & 15);    // fused col 0..191
    int sel = j / 48;
    int jj = j - sel * 48;
    const float* srcs[4] = {sw, qw, kw2, vw};
    dst[idx] = (_Float16)srcs[sel][(size_t)c * 48 + jj];
}

// generic conv weight pack: OIHW -> [R][KS][NT][32 lanes][16 e] f16
__global__ void pack_conv_kernel(const float* __restrict__ src, int Cout, int Cin, int R,
                                 int NT, int KS, _Float16* __restrict__ dst, int total) {
    int idx = blockIdx.x * 256 + threadIdx.x;
    if (idx >= total) return;
    int e = idx & 15, lane = (idx >> 4) & 31;
    int rest = idx >> 9;
    int nt = rest % NT;
    int rest2 = rest / NT;
    int ks = rest2 % KS;
    int r = rest2 / KS;
    int K = bkidx(e, lane);
    int inc = ks * 32 + K;
    int outc = nt * 16 + (lane & 15);
    float v = (inc < Cin && outc < Cout) ? src[((size_t)outc * Cin + inc) * R + r] : 0.f;
    dst[idx] = (_Float16)v;
}

// ============ fused LayerNorm + 4 projections (WMMA) ============
// block: 128 thr (4 waves), 16 rows; wave w owns cols [48w, 48w+48)
__global__ void ln_proj_kernel(const float* __restrict__ x, const float* __restrict__ g,
                               const float* __restrict__ bta, const float* __restrict__ lnsg,
                               const float* __restrict__ lnsb, const float* __restrict__ skipb,
                               const float* __restrict__ qb, const float* __restrict__ kb,
                               const float* __restrict__ vb, const _Float16* __restrict__ wf,
                               _Float16* __restrict__ qh, _Float16* __restrict__ kh,
                               _Float16* __restrict__ vh, float* __restrict__ skip1) {
    __shared__ _Float16 xs[16 * 1024];  // 32 KB: xn tile in f16
    const int b = blockIdx.y;
    const int row0 = blockIdx.x * 16;
    const int tid = threadIdx.x, lane = tid & 31, wave = tid >> 5;

    // ---- LayerNorm over D=1024: 8 threads per row ----
    const int r = tid >> 3, s = tid & 7;
    const float* xr = x + ((size_t)(b * 4096 + row0 + r)) * 1024;
    float sum = 0.f, sq = 0.f;
#pragma unroll 4
    for (int k = 0; k < 128; ++k) {
        float v = xr[s + 8 * k];
        sum += v; sq += v * v;
    }
#pragma unroll
    for (int m = 1; m < 8; m <<= 1) {
        sum += __shfl_xor(sum, m, 32);
        sq  += __shfl_xor(sq,  m, 32);
    }
    float mean = sum * (1.f / 1024.f);
    float var  = sq  * (1.f / 1024.f) - mean * mean;
    float inv  = rsqrtf(var + EPS);
#pragma unroll 4
    for (int k = 0; k < 128; ++k) {
        int c = s + 8 * k;
        xs[r * 1024 + c] = (_Float16)((xr[c] - mean) * inv * g[c] + bta[c]);
    }
    __syncthreads();

    // ---- GEMM: [16 x 1024] * [1024 x 192], wave w -> n-tiles {3w,3w+1,3w+2} ----
    v8f a0 = vz8(), a1 = vz8(), a2 = vz8();
    const int col16 = lane & 15, hi8 = (lane & 16) ? 8 : 0;
    for (int ks = 0; ks < 32; ++ks) {
        v16h av;
#pragma unroll
        for (int e = 0; e < 16; ++e) av[e] = xs[col16 * 1024 + ks * 32 + akidx(e, lane)];
        v16h b0 = *(const v16h*)(wf + ((size_t)(ks * 12 + 3 * wave + 0) * 512 + lane * 16));
        a0 = wmma32(av, b0, a0);
        v16h b1 = *(const v16h*)(wf + ((size_t)(ks * 12 + 3 * wave + 1) * 512 + lane * 16));
        a1 = wmma32(av, b1, a1);
        v16h b2 = *(const v16h*)(wf + ((size_t)(ks * 12 + 3 * wave + 2) * 512 + lane * 16));
        a2 = wmma32(av, b2, a2);
    }

    if (wave == 0) {
        // skip branch: bias + second LN over K=48, fp32 out
#pragma unroll
        for (int vg = 0; vg < 8; ++vg) {
            float v0 = a0[vg] + skipb[col16];
            float v1 = a1[vg] + skipb[16 + col16];
            float v2 = a2[vg] + skipb[32 + col16];
            float t  = v0 + v1 + v2;
            float q2 = v0 * v0 + v1 * v1 + v2 * v2;
#pragma unroll
            for (int m = 1; m < 16; m <<= 1) {
                t  += __shfl_xor(t,  m, 32);
                q2 += __shfl_xor(q2, m, 32);
            }
            float mn = t * (1.f / 48.f);
            float vr = q2 * (1.f / 48.f) - mn * mn;
            float iv = rsqrtf(vr + EPS);
            int rowg = row0 + vg + hi8;
            float* o = skip1 + ((size_t)(b * 4096) + rowg) * 48;
            o[col16]      = (v0 - mn) * iv * lnsg[col16]      + lnsb[col16];
            o[16 + col16] = (v1 - mn) * iv * lnsg[16 + col16] + lnsb[16 + col16];
            o[32 + col16] = (v2 - mn) * iv * lnsg[32 + col16] + lnsb[32 + col16];
        }
    } else {
        const float* bb = (wave == 1) ? qb : (wave == 2 ? kb : vb);
        _Float16* dst   = (wave == 1) ? qh : (wave == 2 ? kh : vh);
#pragma unroll
        for (int vg = 0; vg < 8; ++vg) {
            int rowg = row0 + vg + hi8;
            _Float16* o = dst + ((size_t)(b * 4096) + rowg) * 48;
            o[col16]      = (_Float16)(a0[vg] + bb[col16]);
            o[16 + col16] = (_Float16)(a1[vg] + bb[16 + col16]);
            o[32 + col16] = (_Float16)(a2[vg] + bb[32 + col16]);
        }
    }
}

// ============ flash attention (unscaled softmax) + skip, WMMA ============
// block: 128 thr (4 waves), each wave one 16-query tile; streams 256 key tiles
__global__ void attn_kernel(const _Float16* __restrict__ qh, const _Float16* __restrict__ kh,
                            const _Float16* __restrict__ vh, const float* __restrict__ skip1,
                            float* __restrict__ sem) {
    __shared__ _Float16 Vt[16 * 48];   // V tile, [key][c]
    __shared__ _Float16 Pt[4][256];    // per-wave P tile, [row][key]
    const int b = blockIdx.y;
    const int tid = threadIdx.x, lane = tid & 31, wave = tid >> 5;
    const int row0 = blockIdx.x * 64 + wave * 16;
    const int col16 = lane & 15, hi8 = (lane & 16) ? 8 : 0;

    // Q as A operand (K-dim 48 padded to 64 -> 2 k-steps)
    v16h aq0, aq1;
    {
        const _Float16* qr = qh + ((size_t)(b * 4096) + row0 + col16) * 48;
#pragma unroll
        for (int e = 0; e < 16; ++e) {
            int k0 = akidx(e, lane);
            aq0[e] = qr[k0];
            int c1 = 32 + k0;
            aq1[e] = (c1 < 48) ? qr[c1] : (_Float16)0.f;
        }
    }
    v8f o0 = vz8(), o1 = vz8(), o2 = vz8();
    float rm[8], rl[8];
#pragma unroll
    for (int i = 0; i < 8; ++i) { rm[i] = -1e30f; rl[i] = 0.f; }

    const _Float16* kbase = kh + (size_t)(b * 4096) * 48;
    const _Float16* vbase = vh + (size_t)(b * 4096) * 48;

    for (int j = 0; j < 256; ++j) {
        for (int t = tid; t < 768; t += 128) Vt[t] = vbase[(size_t)(j * 16) * 48 + t];
        __syncthreads();
        if (j + 1 < 256) __builtin_prefetch(kbase + (size_t)((j + 1) * 16) * 48, 0, 1);

        // S = Q @ K^T : B operand = contiguous 32B per lane from one K row
        const _Float16* krow = kbase + (size_t)(j * 16 + col16) * 48;
        v16h bk0 = *(const v16h*)(krow + (lane & 16));
        v16h bk1 = (lane & 16) ? vz16() : *(const v16h*)(krow + 32);
        v8f sv8 = vz8();
        sv8 = wmma32(aq0, bk0, sv8);
        sv8 = wmma32(aq1, bk1, sv8);

        // online softmax update per row
#pragma unroll
        for (int vg = 0; vg < 8; ++vg) {
            float sv = sv8[vg];
            float mx = sv;
#pragma unroll
            for (int m = 1; m < 16; m <<= 1) mx = fmaxf(mx, __shfl_xor(mx, m, 32));
            float nm = fmaxf(rm[vg], mx);
            float sc = __expf(rm[vg] - nm);
            float p  = __expf(sv - nm);
            float ps = p;
#pragma unroll
            for (int m = 1; m < 16; m <<= 1) ps += __shfl_xor(ps, m, 32);
            rl[vg] = rl[vg] * sc + ps;
            rm[vg] = nm;
            o0[vg] *= sc; o1[vg] *= sc; o2[vg] *= sc;
            Pt[wave][(vg + hi8) * 16 + col16] = (_Float16)p;
        }
        __syncthreads();

        // P (A operand, keys 0..15 valid) @ V (B operand, 3 n-tiles of 16 cols)
        v16h ap;
#pragma unroll
        for (int e = 0; e < 16; ++e) {
            int k0 = akidx(e, lane);
            ap[e] = (k0 < 16) ? Pt[wave][col16 * 16 + k0] : (_Float16)0.f;
        }
        v16h bv0, bv1, bv2;
#pragma unroll
        for (int e = 0; e < 16; ++e) {
            int kk = bkidx(e, lane);
            if (kk < 16) {
                const _Float16* vr = Vt + kk * 48;
                bv0[e] = vr[col16]; bv1[e] = vr[16 + col16]; bv2[e] = vr[32 + col16];
            } else {
                bv0[e] = (_Float16)0.f; bv1[e] = (_Float16)0.f; bv2[e] = (_Float16)0.f;
            }
        }
        o0 = wmma32(ap, bv0, o0);
        o1 = wmma32(ap, bv1, o1);
        o2 = wmma32(ap, bv2, o2);
        __syncthreads();
    }

    // finalize: att = O/l + skip1 ; write channel-major sem [B,48,4096]
#pragma unroll
    for (int vg = 0; vg < 8; ++vg) {
        int rowg = row0 + vg + hi8;
        float iv = 1.f / rl[vg];
        const float* sk = skip1 + ((size_t)(b * 4096) + rowg) * 48;
        float* sb = sem + (size_t)b * 48 * 4096;
        sb[(size_t)(col16) * 4096 + rowg]      = o0[vg] * iv + sk[col16];
        sb[(size_t)(16 + col16) * 4096 + rowg] = o1[vg] * iv + sk[16 + col16];
        sb[(size_t)(32 + col16) * 4096 + rowg] = o2[vg] * iv + sk[32 + col16];
    }
}

// ============ conv1: 3x3 48->128 + BN(eval) + ReLU (WMMA) ============
// block 256 thr (8 waves), 16-pixel strip; wave w -> out-channel tile w
__global__ void conv1_kernel(const float* __restrict__ sem, const _Float16* __restrict__ wp,
                             const float* __restrict__ cb, const float* __restrict__ bng,
                             const float* __restrict__ bnb, const float* __restrict__ bnm,
                             const float* __restrict__ bnv, float* __restrict__ hbuf) {
    __shared__ _Float16 patch[48 * 3 * 18];  // [c][yy][xx], halo included
    const int b = blockIdx.y, t = blockIdx.x;
    const int y = t >> 2, x0 = (t & 3) * 16, pix0 = y * 64 + x0;
    const int tid = threadIdx.x, lane = tid & 31, wave = tid >> 5;
    for (int idx = tid; idx < 2592; idx += 256) {
        int c = idx / 54, rem = idx % 54, yy = rem / 18, xx = rem % 18;
        int gy = y + yy - 1, gx = x0 + xx - 1;
        float v = (gy >= 0 && gy < 64 && gx >= 0 && gx < 64)
                      ? sem[((size_t)(b * 48 + c)) * 4096 + gy * 64 + gx] : 0.f;
        patch[idx] = (_Float16)v;
    }
    __syncthreads();
    const int col16 = lane & 15, hi8 = (lane & 16) ? 8 : 0;
    v8f acc = vz8();
    for (int rr = 0; rr < 9; ++rr) {
        int dy = rr / 3, dx = rr % 3;
        for (int ks = 0; ks < 2; ++ks) {
            v16h av;
#pragma unroll
            for (int e = 0; e < 16; ++e) {
                int c = ks * 32 + akidx(e, lane);
                av[e] = (c < 48) ? patch[c * 54 + dy * 18 + col16 + dx] : (_Float16)0.f;
            }
            v16h bv = *(const v16h*)(wp + ((size_t)((rr * 2 + ks) * 8 + wave) * 512 + lane * 16));
            acc = wmma32(av, bv, acc);
        }
    }
    int ch = wave * 16 + col16;
    float bscale = rsqrtf(bnv[ch] + EPS) * bng[ch];
#pragma unroll
    for (int vg = 0; vg < 8; ++vg) {
        int pix = pix0 + vg + hi8;
        float v = acc[vg] + cb[ch];
        v = (v - bnm[ch]) * bscale + bnb[ch];
        hbuf[((size_t)(b * 128 + ch)) * 4096 + pix] = fmaxf(v, 0.f);
    }
}

// ============ conv2: 1x1 128->48 -> semantic_map (WMMA GEMM) ============
// block 128 thr; waves 0..2 each own one 16-col n-tile
__global__ void conv2_kernel(const float* __restrict__ hbuf, const _Float16* __restrict__ wp,
                             const float* __restrict__ cb, float* __restrict__ smap) {
    __shared__ _Float16 als[128 * 16];  // [c][m]
    const int b = blockIdx.y, pix0 = blockIdx.x * 16;
    const int tid = threadIdx.x, lane = tid & 31, wave = tid >> 5;
    for (int idx = tid; idx < 2048; idx += 128) {
        int c = idx >> 4, m = idx & 15;
        als[idx] = (_Float16)hbuf[((size_t)(b * 128 + c)) * 4096 + pix0 + m];
    }
    __syncthreads();
    if (wave < 3) {
        const int col16 = lane & 15, hi8 = (lane & 16) ? 8 : 0;
        v8f acc = vz8();
        for (int ks = 0; ks < 4; ++ks) {
            v16h av;
#pragma unroll
            for (int e = 0; e < 16; ++e) av[e] = als[(ks * 32 + akidx(e, lane)) * 16 + col16];
            v16h bv = *(const v16h*)(wp + ((size_t)(ks * 3 + wave) * 512 + lane * 16));
            acc = wmma32(av, bv, acc);
        }
        int ch = wave * 16 + col16;
#pragma unroll
        for (int vg = 0; vg < 8; ++vg)
            smap[((size_t)(b * 48 + ch)) * 4096 + pix0 + vg + hi8] = acc[vg] + cb[ch];
    }
}

// ============ convg: 3x3 48->1024 + fused residual: out = x + gamma*seg ============
// block 256 thr (8 waves), wave w -> out-channel tiles 8w..8w+7
__global__ void convg_kernel(const float* __restrict__ smap, const _Float16* __restrict__ wp,
                             const float* __restrict__ cb, const float* __restrict__ x,
                             const float* __restrict__ gamma, float* __restrict__ out) {
    __shared__ _Float16 patch[48 * 3 * 18];
    const int b = blockIdx.y, t = blockIdx.x;
    const int y = t >> 2, x0 = (t & 3) * 16, pix0 = y * 64 + x0;
    const int tid = threadIdx.x, lane = tid & 31, wave = tid >> 5;
    for (int idx = tid; idx < 2592; idx += 256) {
        int c = idx / 54, rem = idx % 54, yy = rem / 18, xx = rem % 18;
        int gy = y + yy - 1, gx = x0 + xx - 1;
        float v = (gy >= 0 && gy < 64 && gx >= 0 && gx < 64)
                      ? smap[((size_t)(b * 48 + c)) * 4096 + gy * 64 + gx] : 0.f;
        patch[idx] = (_Float16)v;
    }
    __syncthreads();
    const int col16 = lane & 15, hi8 = (lane & 16) ? 8 : 0;
    v8f acc[8];
#pragma unroll
    for (int i = 0; i < 8; ++i) acc[i] = vz8();
    for (int rr = 0; rr < 9; ++rr) {
        int dy = rr / 3, dx = rr % 3;
        for (int ks = 0; ks < 2; ++ks) {
            v16h av;
#pragma unroll
            for (int e = 0; e < 16; ++e) {
                int c = ks * 32 + akidx(e, lane);
                av[e] = (c < 48) ? patch[c * 54 + dy * 18 + col16 + dx] : (_Float16)0.f;
            }
#pragma unroll
            for (int tt = 0; tt < 8; ++tt) {
                int nt = wave * 8 + tt;
                v16h bv = *(const v16h*)(wp + ((size_t)((rr * 2 + ks) * 64 + nt) * 512 + lane * 16));
                acc[tt] = wmma32(av, bv, acc[tt]);
            }
        }
    }
    float gm = gamma[0];
#pragma unroll
    for (int tt = 0; tt < 8; ++tt) {
        int ch = (wave * 8 + tt) * 16 + col16;
        float bias = cb[ch];
#pragma unroll
        for (int vg = 0; vg < 8; ++vg) {
            int pix = pix0 + vg + hi8;
            size_t o = ((size_t)(b * 4096) + pix) * 1024 + ch;
            out[o] = x[o] + gm * (acc[tt][vg] + bias);
        }
    }
}

// ============================ host launcher ============================
extern "C" void kernel_launch(void* const* d_in, const int* in_sizes, int n_in,
                              void* d_out, int out_size, void* d_ws, size_t ws_size,
                              hipStream_t stream) {
    const float* x      = (const float*)d_in[0];
    const float* ln1_g  = (const float*)d_in[1];
    const float* ln1_b  = (const float*)d_in[2];
    const float* lns_g  = (const float*)d_in[3];
    const float* lns_b  = (const float*)d_in[4];
    const float* skip_w = (const float*)d_in[5];
    const float* skip_b = (const float*)d_in[6];
    const float* qw     = (const float*)d_in[7];
    const float* qb     = (const float*)d_in[8];
    const float* kw     = (const float*)d_in[9];
    const float* kb     = (const float*)d_in[10];
    const float* vw     = (const float*)d_in[11];
    const float* vb     = (const float*)d_in[12];
    const float* conv1w = (const float*)d_in[13];
    const float* conv1b = (const float*)d_in[14];
    const float* bn_g   = (const float*)d_in[15];
    const float* bn_b   = (const float*)d_in[16];
    const float* bn_m   = (const float*)d_in[17];
    const float* bn_v   = (const float*)d_in[18];
    const float* conv2w = (const float*)d_in[19];
    const float* conv2b = (const float*)d_in[20];
    const float* convgw = (const float*)d_in[21];
    const float* convgb = (const float*)d_in[22];
    const float* gamma  = (const float*)d_in[23];

    char* ws = (char*)d_ws;
    size_t off = 0;
    auto alloc = [&](size_t bytes) -> void* {
        void* p = ws + off;
        off = (off + bytes + 255) & ~(size_t)255;
        return p;
    };
    _Float16* wf   = (_Float16*)alloc((size_t)32 * 12 * 512 * 2);  // 384 KB
    _Float16* wc1  = (_Float16*)alloc((size_t)9 * 2 * 8 * 512 * 2);
    _Float16* wcg  = (_Float16*)alloc((size_t)9 * 2 * 64 * 512 * 2);
    _Float16* wc2  = (_Float16*)alloc((size_t)4 * 3 * 512 * 2);
    _Float16* qh   = (_Float16*)alloc((size_t)4 * 4096 * 48 * 2);
    _Float16* kh   = (_Float16*)alloc((size_t)4 * 4096 * 48 * 2);
    _Float16* vh   = (_Float16*)alloc((size_t)4 * 4096 * 48 * 2);
    float*    skip1 = (float*)alloc((size_t)4 * 4096 * 48 * 4);
    float*    sem   = (float*)alloc((size_t)4 * 48 * 4096 * 4);
    float*    hbuf  = (float*)alloc((size_t)4 * 128 * 4096 * 4);

    float* out  = (float*)d_out;                       // [B,N,D]
    float* smap = out + (size_t)4 * 4096 * 1024;       // [B,48,H,W]

    // 1) pack weights (WMMA B-operand layout, f16)
    pack_proj_kernel<<<(32 * 12 * 512 + 255) / 256, 256, 0, stream>>>(skip_w, qw, kw, vw, wf);
    {
        int tot1 = 9 * 2 * 8 * 512;
        pack_conv_kernel<<<(tot1 + 255) / 256, 256, 0, stream>>>(conv1w, 128, 48, 9, 8, 2, wc1, tot1);
        int totg = 9 * 2 * 64 * 512;
        pack_conv_kernel<<<(totg + 255) / 256, 256, 0, stream>>>(convgw, 1024, 48, 9, 64, 2, wcg, totg);
        int tot2 = 4 * 3 * 512;
        pack_conv_kernel<<<(tot2 + 255) / 256, 256, 0, stream>>>(conv2w, 48, 128, 1, 3, 4, wc2, tot2);
    }
    // 2) LN + projections (+ second LN on skip)
    ln_proj_kernel<<<dim3(256, 4), 128, 0, stream>>>(x, ln1_g, ln1_b, lns_g, lns_b, skip_b,
                                                     qb, kb, vb, wf, qh, kh, vh, skip1);
    // 3) flash attention + skip -> sem (NCHW)
    attn_kernel<<<dim3(64, 4), 128, 0, stream>>>(qh, kh, vh, skip1, sem);
    // 4) conv1 3x3 + BN + ReLU
    conv1_kernel<<<dim3(256, 4), 256, 0, stream>>>(sem, wc1, conv1b, bn_g, bn_b, bn_m, bn_v, hbuf);
    // 5) conv2 1x1 -> semantic_map (output #2)
    conv2_kernel<<<dim3(256, 4), 128, 0, stream>>>(hbuf, wc2, conv2b, smap);
    // 6) convg 3x3 + fused residual -> out (output #1)
    convg_kernel<<<dim3(256, 4), 256, 0, stream>>>(smap, wcg, convgb, x, gamma, out);
}
